// DeepSetLayer1_59459527246448
// MI455X (gfx1250) — matrix operand order repaired and checked by reading, not compiled, not run on hardware
//
#include <hip/hip_runtime.h>

typedef __attribute__((ext_vector_type(16))) _Float16 v16h;
typedef __attribute__((ext_vector_type(8)))  _Float16 v8h;
typedef __attribute__((ext_vector_type(4)))  _Float16 v4h;
typedef __attribute__((ext_vector_type(8)))  float    v8f;

#define INC  128
#define OUTC 128
#define TILE_ROWS 128
#define NTILES 4            // row-tiles processed per block
#define NHALF 16384         // halves per 128x128 matrix

// ---- dynamic LDS layout (bytes), MLP kernel ----
#define OFF_W1   0
#define OFF_W2   32768
#define OFF_X    65536      // A-swizzled x tile; reused as GEMM2 output tile
#define OFF_H    98304      // A-swizzled h tile
#define OFF_B1   131072
#define OFF_B2   131584
#define OFF_SEG  132096
#define SMEM_MLP 132608

// ---- dynamic LDS layout (bytes), final kernel ----
#define OFF_W3   0
#define OFF_M    32768
#define OFF_B3   65536
#define SMEM_FIN 66048

// Fragment-resident layouts (all f16):
//  A-operand buffer: [mb(8)][kb(4)][part(2)][lane(32)][sub(8)]  (x / h tiles)
//   element (row=16*mb+m, col=32*kb+kk):  half=(kk>>3)&1, lane=m|(half<<4),
//                                         part=(kk>>4)&1, sub=kk&7
//  B-operand buffer: [kb(4)][nb(8)][part(2)][lane(32)][sub(8)]  (weights)
//   element (k=32*kb+kk, n=16*nb+nn):     lane=nn|(((kk>>3)&1)<<4),
//                                         part=(kk>>4)&1, sub=kk&7

__device__ __forceinline__ v16h frag_ld(const _Float16* frag_base) {
  const int lane = threadIdx.x & 31;
  const _Float16* p = frag_base + lane * 8;
  v8h lo = *(const v8h*)p;          // part 0 -> one ds_load_b128
  v8h hi = *(const v8h*)(p + 256);  // part 1 -> one ds_load_b128
  v16h a;
#pragma unroll
  for (int j = 0; j < 8; ++j) { a[j] = lo[j]; a[8 + j] = hi[j]; }
  return a;
}

__device__ __forceinline__ v16h frag_a(const _Float16* buf, int mb, int kb) {
  return frag_ld(buf + ((mb * 4 + kb) * 2) * 256);
}
__device__ __forceinline__ v16h frag_b(const _Float16* buf, int kb, int nb) {
  return frag_ld(buf + ((kb * 8 + nb) * 2) * 256);
}

__device__ __forceinline__ v8f wmma_f16(v16h a, v16h b, v8f c) {
  return __builtin_amdgcn_wmma_f32_16x16x32_f16(false, a, false, b, (short)0, c, false, false);
}

// ---- one-time weight swizzle: row-major f32 -> B-fragment-layout f16 ----
__global__ void __launch_bounds__(256)
deepset_prep_weights(const float* __restrict__ W1, const float* __restrict__ W2,
                     const float* __restrict__ W3, _Float16* __restrict__ w1s,
                     _Float16* __restrict__ w2s, _Float16* __restrict__ w3s) {
  int i = blockIdx.x * 256 + threadIdx.x;      // 0 .. 16383
  if (i >= NHALF) return;
  int k = i >> 7, n = i & 127;
  int kb = k >> 5, kk = k & 31;
  int nb = n >> 4, nn = n & 15;
  int lane = nn | (((kk >> 3) & 1) << 4);
  int part = (kk >> 4) & 1;
  int sub  = kk & 7;
  int o = (((kb * 8 + nb) * 2 + part) * 32 + lane) * 8 + sub;
  w1s[o] = (_Float16)W1[i];
  w2s[o] = (_Float16)W2[i];
  w3s[o] = (_Float16)W3[i];
}

__global__ void deepset_zero_kernel(float* __restrict__ p, int n) {
  int i = blockIdx.x * blockDim.x + threadIdx.x;
  if (i < n) p[i] = 0.0f;
}

__global__ void __launch_bounds__(256)
deepset_mlp_seg_kernel(const float* __restrict__ x1, const int* __restrict__ es,
                       const _Float16* __restrict__ w1s, const _Float16* __restrict__ w2s,
                       const float* __restrict__ b1, const float* __restrict__ b2,
                       float* __restrict__ sums, long long E, int nB) {
  extern __shared__ char smem[];
  _Float16* lw1 = (_Float16*)(smem + OFF_W1);
  _Float16* lw2 = (_Float16*)(smem + OFF_W2);
  _Float16* lx  = (_Float16*)(smem + OFF_X);   // x tile, later GEMM2 output tile
  _Float16* lh  = (_Float16*)(smem + OFF_H);
  float*    lb1 = (float*)(smem + OFF_B1);
  float*    lb2 = (float*)(smem + OFF_B2);
  int*      lseg = (int*)(smem + OFF_SEG);

  const int t = threadIdx.x;

  // stage pre-swizzled weights: contiguous 16B copies (b128 load + b128 store)
  {
    const uint4* g1 = (const uint4*)w1s;
    const uint4* g2 = (const uint4*)w2s;
    uint4* s1 = (uint4*)lw1;
    uint4* s2 = (uint4*)lw2;
#pragma unroll
    for (int i = t; i < NHALF / 8; i += 256) { s1[i] = g1[i]; s2[i] = g2[i]; }
  }
  if (t < OUTC) { lb1[t] = b1[t]; lb2[t] = b2[t]; }

  const int lane  = t & 31;
  const int wave  = t >> 5;          // 8 waves, each owns a 16-row strip (mb = wave)
  const int ncol  = lane & 15;
  const int rbase = (lane >> 4) << 3;

  for (int tile = 0; tile < NTILES; ++tile) {
    const long long rowbase =
        ((long long)blockIdx.x * NTILES + tile) * TILE_ROWS;
    __syncthreads();   // previous iteration's reads of lx/lseg complete

    // stage x tile: f32 float4 loads -> f16, written in A-fragment layout
    for (int i = t; i < TILE_ROWS * INC / 4; i += 256) {
      int r  = i >> 5;               // 32 float4 per row
      int c4 = (i & 31) * 4;
      long long gr = rowbase + r;
      float4 f = make_float4(0.f, 0.f, 0.f, 0.f);
      if (gr < E) f = ((const float4*)x1)[gr * (INC / 4) + (i & 31)];
      int mb = r >> 4, m = r & 15;
      int kb = c4 >> 5, kk = c4 & 31;
      int half = (kk >> 3) & 1, part = (kk >> 4) & 1, sub = kk & 7;
      int lane2 = m | (half << 4);
      v4h h4; h4[0] = (_Float16)f.x; h4[1] = (_Float16)f.y;
              h4[2] = (_Float16)f.z; h4[3] = (_Float16)f.w;
      *(v4h*)&lx[(((mb * 4 + kb) * 2 + part) * 32 + lane2) * 8 + sub] = h4;
    }

    // per-row segment id: searchsorted(edge_slices[1:], row, 'right')
    if (t < TILE_ROWS) {
      long long gr = rowbase + t;
      int s = -1;
      if (gr < E) {
        int lo = 0, hi = nB;
        while (lo < hi) {
          int mid = (lo + hi) >> 1;
          if ((long long)es[mid + 1] <= gr) lo = mid + 1; else hi = mid;
        }
        s = lo;
      }
      lseg[t] = s;
    }
    __syncthreads();

    // ---- GEMM1: h = relu(x @ W1 + b1), stored in A-fragment layout ----
    {
      v16h a0 = frag_a(lx, wave, 0), a1 = frag_a(lx, wave, 1),
           a2 = frag_a(lx, wave, 2), a3 = frag_a(lx, wave, 3);
#pragma unroll
      for (int nb = 0; nb < 8; ++nb) {
        v8f c = {};
        c = wmma_f16(a0, frag_b(lw1, 0, nb), c);
        c = wmma_f16(a1, frag_b(lw1, 1, nb), c);
        c = wmma_f16(a2, frag_b(lw1, 2, nb), c);
        c = wmma_f16(a3, frag_b(lw1, 3, nb), c);
        float bv = lb1[nb * 16 + ncol];
        // dest in A-layout: kb=nb>>1, part=nb&1, half=(ncol>>3)&1, sub=ncol&7
        int half = (ncol >> 3) & 1, sub = ncol & 7;
        _Float16* dst = lh + (((wave * 4 + (nb >> 1)) * 2 + (nb & 1)) * 32) * 8 + sub;
#pragma unroll
        for (int r = 0; r < 8; ++r) {
          float v = fmaxf(c[r] + bv, 0.0f);
          dst[((rbase + r) | (half << 4)) * 8] = (_Float16)v;
        }
      }
    }
    __syncthreads();   // h complete; x tile region free

    // ---- GEMM2: h @ W2 + b2 -> f16 into lx (A-fragment layout) ----
    {
      v16h a0 = frag_a(lh, wave, 0), a1 = frag_a(lh, wave, 1),
           a2 = frag_a(lh, wave, 2), a3 = frag_a(lh, wave, 3);
#pragma unroll
      for (int nb = 0; nb < 8; ++nb) {
        v8f c = {};
        c = wmma_f16(a0, frag_b(lw2, 0, nb), c);
        c = wmma_f16(a1, frag_b(lw2, 1, nb), c);
        c = wmma_f16(a2, frag_b(lw2, 2, nb), c);
        c = wmma_f16(a3, frag_b(lw2, 3, nb), c);
        float bv = lb2[nb * 16 + ncol];
        int half = (ncol >> 3) & 1, sub = ncol & 7;
        _Float16* dst = lx + (((wave * 4 + (nb >> 1)) * 2 + (nb & 1)) * 32) * 8 + sub;
#pragma unroll
        for (int r = 0; r < 8; ++r)
          dst[((rbase + r) | (half << 4)) * 8] = (_Float16)(c[r] + bv);
      }
    }
    __syncthreads();

    // ---- fused segmented column reduction; few atomic flushes per block ----
    {
      const int col = t & 127;
      const int r0  = (t >> 7) * 64;  // 2 threads per column, 64 rows each
      const int kb = col >> 5, kk = col & 31;
      const int half = (kk >> 3) & 1, part = (kk >> 4) & 1, sub = kk & 7;
      float acc = 0.0f;
      int cur = -2;
#pragma unroll 1
      for (int r = 0; r < 64; ++r) {
        int rr = r0 + r;
        int s = lseg[rr];
        if (s != cur) {
          if (cur >= 0) unsafeAtomicAdd(&sums[(size_t)cur * OUTC + col], acc);
          acc = 0.0f; cur = s;
        }
        int mb = rr >> 4, m = rr & 15;
        acc += (float)lx[((((mb * 4 + kb) * 2 + part) * 32) + (m | (half << 4))) * 8 + sub];
      }
      if (cur >= 0) unsafeAtomicAdd(&sums[(size_t)cur * OUTC + col], acc);
    }
  }
}

__global__ void __launch_bounds__(256)
deepset_mean_out_kernel(const float* __restrict__ sums, const int* __restrict__ es,
                        const _Float16* __restrict__ w3s, const float* __restrict__ b3,
                        float* __restrict__ out, int nB) {
  extern __shared__ char smem[];
  _Float16* lw3 = (_Float16*)(smem + OFF_W3);
  _Float16* lm  = (_Float16*)(smem + OFF_M);
  float*    lb3 = (float*)(smem + OFF_B3);

  const int t  = threadIdx.x;
  const int rb = blockIdx.x * 128;

  {
    const uint4* g3 = (const uint4*)w3s;
    uint4* s3 = (uint4*)lw3;
#pragma unroll
    for (int i = t; i < NHALF / 8; i += 256) s3[i] = g3[i];
  }
  if (t < OUTC) lb3[t] = b3[t];

  // mean tile = sums / max(count,1), written in A-fragment layout
  for (int i = t; i < 128 * OUTC / 4; i += 256) {
    int r  = i >> 5;
    int c4 = (i & 31) * 4;
    int row = rb + r;
    float4 s = make_float4(0.f, 0.f, 0.f, 0.f);
    float inv = 0.0f;
    if (row < nB) {
      s = ((const float4*)sums)[(size_t)row * (OUTC / 4) + (i & 31)];
      inv = 1.0f / fmaxf((float)(es[row + 1] - es[row]), 1.0f);
    }
    int mb = r >> 4, m = r & 15;
    int kb = c4 >> 5, kk = c4 & 31;
    int half = (kk >> 3) & 1, part = (kk >> 4) & 1, sub = kk & 7;
    int lane2 = m | (half << 4);
    v4h h4; h4[0] = (_Float16)(s.x * inv); h4[1] = (_Float16)(s.y * inv);
            h4[2] = (_Float16)(s.z * inv); h4[3] = (_Float16)(s.w * inv);
    *(v4h*)&lm[(((mb * 4 + kb) * 2 + part) * 32 + lane2) * 8 + sub] = h4;
  }
  __syncthreads();

  const int lane  = t & 31;
  const int wave  = t >> 5;
  const int ncol  = lane & 15;
  const int rbase = (lane >> 4) << 3;

  v16h a0 = frag_a(lm, wave, 0), a1 = frag_a(lm, wave, 1),
       a2 = frag_a(lm, wave, 2), a3 = frag_a(lm, wave, 3);
#pragma unroll
  for (int nb = 0; nb < 8; ++nb) {
    v8f c = {};
    c = wmma_f16(a0, frag_b(lw3, 0, nb), c);
    c = wmma_f16(a1, frag_b(lw3, 1, nb), c);
    c = wmma_f16(a2, frag_b(lw3, 2, nb), c);
    c = wmma_f16(a3, frag_b(lw3, 3, nb), c);
    float bv = lb3[nb * 16 + ncol];
#pragma unroll
    for (int r = 0; r < 8; ++r) {
      int row = rb + wave * 16 + rbase + r;
      if (row < nB) out[(size_t)row * OUTC + nb * 16 + ncol] = c[r] + bv;
    }
  }
}

extern "C" void kernel_launch(void* const* d_in, const int* in_sizes, int n_in,
                              void* d_out, int out_size, void* d_ws, size_t ws_size,
                              hipStream_t stream) {
  const float* x1 = (const float*)d_in[0];
  const int*   es = (const int*)  d_in[1];
  const float* W1 = (const float*)d_in[2];
  const float* b1 = (const float*)d_in[3];
  const float* W2 = (const float*)d_in[4];
  const float* b2 = (const float*)d_in[5];
  const float* W3 = (const float*)d_in[6];
  const float* b3 = (const float*)d_in[7];
  float* out = (float*)d_out;

  const long long E  = (long long)in_sizes[0] / INC;
  const int       nB = in_sizes[1] - 1;

  // workspace: [sums: nB*OUTC f32][w1s|w2s|w3s: 16384 f16 each]
  float* sums = (float*)d_ws;
  _Float16* w1s = (_Float16*)((char*)d_ws + (size_t)nB * OUTC * sizeof(float));
  _Float16* w2s = w1s + NHALF;
  _Float16* w3s = w2s + NHALF;

  const int nsum = nB * OUTC;
  deepset_zero_kernel<<<(nsum + 255) / 256, 256, 0, stream>>>(sums, nsum);
  deepset_prep_weights<<<NHALF / 256, 256, 0, stream>>>(W1, W2, W3, w1s, w2s, w3s);

  const long long rows_per_block = (long long)TILE_ROWS * NTILES;
  const int nblk = (int)((E + rows_per_block - 1) / rows_per_block);
  deepset_mlp_seg_kernel<<<nblk, 256, SMEM_MLP, stream>>>(x1, es, w1s, w2s, b1, b2,
                                                          sums, E, nB);

  deepset_mean_out_kernel<<<(nB + 127) / 128, 256, SMEM_FIN, stream>>>(sums, es, w3s,
                                                                       b3, out, nB);
}
